// InfiniAttentionLayer_53987738911175
// MI455X (gfx1250) — compile-verified, matrix operand order checked by reading, not run on hardware
//
#include <hip/hip_runtime.h>
#include <hip/hip_bf16.h>
#include <stdint.h>

// ---------------------------------------------------------------------------
// InfiniAttention layer for MI455X (gfx1250, wave32, WMMA).
// All four big GEMMs (Q/K/V proj, retrieve, memory update, out proj) run on
// v_wmma_f32_16x16x32_bf16 with fp32 accumulation. Main-loop staging uses
// CDNA5 async global->LDS copies (ASYNCcnt) with LDS double buffering so the
// next K-slab streams in underneath the current slab's WMMAs. Epilogues fuse
// phi/elu, norm division, M accumulation, residual mix; transposed scatter
// epilogues are coalesced through an LDS transpose.
// ---------------------------------------------------------------------------

typedef __bf16 bf16_t;
typedef __attribute__((ext_vector_type(16))) __bf16 v16bf;
typedef __attribute__((ext_vector_type(8)))  __bf16 v8bf;
typedef __attribute__((ext_vector_type(8)))  float  v8f;

union V16U { v16bf v; v8bf h[2]; };

__device__ __forceinline__ bf16_t f2bf(float f) {
    union { float f; unsigned u; } x{f};
    unsigned r = x.u + 0x7FFFu + ((x.u >> 16) & 1u);   // round-to-nearest-even
    unsigned short h = (unsigned short)(r >> 16);
    return __builtin_bit_cast(bf16_t, h);
}

__device__ __forceinline__ float phif(float x) {      // elu(x)+1
    return x > 0.0f ? x + 1.0f : __expf(x);
}

// CDNA5 async copy: 16B global -> LDS, tracked by ASYNCcnt (ISA 15.18.3 op 98).
// LDS operand is the wave-relative LDS byte offset = low 32 bits of the
// generic pointer to a __shared__ object (flat LDS aperture: addr[31:0]).
__device__ __forceinline__ void async_copy_b128(const void* g, void* l) {
    asm volatile("global_load_async_to_lds_b128 %0, %1, off"
                 :: "v"((uint32_t)(uintptr_t)l), "v"((uint64_t)(uintptr_t)g)
                 : "memory");
}
__device__ __forceinline__ void wait_async_le4() {
    asm volatile("s_wait_asynccnt 0x4" ::: "memory");
}
__device__ __forceinline__ void wait_async_0() {
    asm volatile("s_wait_asynccnt 0x0" ::: "memory");
}

// ---------------------------------------------------------------------------
// Generic bf16 WMMA GEMM: C(M x N) = A(M x K) @ B(N x K)^T, fp32 accum.
// Block tile 128x128, 256 threads = 8 wave32 waves, each wave owns 32x64
// (2x4 WMMA fragments). K is consumed in 32-wide slabs, double buffered in
// LDS and filled by async global->LDS B128 copies.
// MODE epilogues:
//  1: Obf = phi(acc)                      (phiQ, natural [m][n] layout)
//  2: Obf[phiK^T scatter] = phi(acc)      ([b][h][d][t], LDS-transposed store)
//  3: Obf[V^T scatter]    = acc           ([b][h][d][t], LDS-transposed store)
//  4: Obf = acc / max(norm[m],1e-6)       (A_mem -> mem_flat)
//  5: Of32 = aux0 + acc                   (M_new = M + update)
//  6: Of32 = aux0 + scalar[0]*acc         (mixed = hs + beta*mem_out)
// z-batching: per-z offsets = (z>>4)*zhi + (z&15)*zlo.
// ---------------------------------------------------------------------------
template<int MODE>
__global__ __launch_bounds__(256)
void gemm_wmma_bf16(const bf16_t* __restrict__ A,  long lda, long a_zhi, long a_zlo,
                    const bf16_t* __restrict__ Bm, long ldb, long b_zhi, long b_zlo,
                    int Ktot,
                    bf16_t* __restrict__ Obf, float* __restrict__ Of32,
                    long ldo, long o_zhi, long o_zlo,
                    const float* __restrict__ aux0, long x_zhi, long x_zlo,
                    const float* __restrict__ scalarp)
{
    // 32KB: A slabs [2][128][32] at 0..8191, B slabs [2][128][32] at 8192..16383.
    // Reused after the main loop as a 128x128 bf16 transpose buffer (modes 2/3).
    __shared__ __align__(16) bf16_t smem[16384];

    const int z  = blockIdx.z;
    const int zh = z >> 4, zl = z & 15;
    A  += (long)zh * a_zhi + (long)zl * a_zlo;
    Bm += (long)zh * b_zhi + (long)zl * b_zlo;
    if (aux0) aux0 += (long)zh * x_zhi + (long)zl * x_zlo;
    const long o_off = (long)zh * o_zhi + (long)zl * o_zlo;

    const int m0 = blockIdx.y * 128;
    const int n0 = blockIdx.x * 128;

    const int tid  = threadIdx.x;
    const int lrow = tid >> 1;           // 0..127 : tile row staged by this thread
    const int lseg = (tid & 1) << 4;     // 0 or 16: 16-element K segment

    const int wave = tid >> 5;           // 8 wave32 waves
    const int lane = tid & 31;
    const int wm   = (wave & 3) * 32;    // wave tile row
    const int wn   = (wave >> 2) * 64;   // wave tile col
    const int half = lane >> 4;          // K-half select per WMMA layout
    const int l16  = lane & 15;

    const bf16_t* gA = A  + (long)(m0 + lrow) * lda + lseg;   // + k0 later
    const bf16_t* gB = Bm + (long)(n0 + lrow) * ldb + lseg;
    bf16_t* lA = &smem[lrow * 32 + lseg];                     // + buf*4096
    bf16_t* lB = &smem[8192 + lrow * 32 + lseg];

    const v8f vzero = {0.f, 0.f, 0.f, 0.f, 0.f, 0.f, 0.f, 0.f};
    v8f acc[2][4];
    for (int i = 0; i < 2; ++i)
        for (int j = 0; j < 4; ++j)
            acc[i][j] = vzero;

    const int nslab = Ktot >> 5;

    // prologue: stage slab 0 into buffer 0 (4 async B128 per thread)
    async_copy_b128(gA,     lA);
    async_copy_b128(gA + 8, lA + 8);
    async_copy_b128(gB,     lB);
    async_copy_b128(gB + 8, lB + 8);

    for (int s = 0; s < nslab; ++s) {
        const int cur = s & 1;
        if (s + 1 < nslab) {
            // stage slab s+1 into the other buffer while slab s computes
            const int nk  = (s + 1) << 5;
            const int nb  = (cur ^ 1) * 4096;
            async_copy_b128(gA + nk,     lA + nb);
            async_copy_b128(gA + nk + 8, lA + nb + 8);
            async_copy_b128(gB + nk,     lB + nb);
            async_copy_b128(gB + nk + 8, lB + nb + 8);
            if (s + 2 < nslab) {                     // L2 prefetch of slab s+2
                __builtin_prefetch(gA + nk + 32, 0, 1);
                __builtin_prefetch(gB + nk + 32, 0, 1);
            }
            wait_async_le4();   // slab s landed; slab s+1 still in flight
        } else {
            wait_async_0();
        }
        __syncthreads();

        // 16-bit fragment layout: lane holds row (|| col) l16,
        // K = half*8..+7 in dwords 0..3 and 16+half*8..+7 in dwords 4..7.
        const int abase = cur * 4096;
        V16U af[2];
        for (int i = 0; i < 2; ++i) {
            const bf16_t* ar = &smem[abase + (wm + i * 16 + l16) * 32 + half * 8];
            af[i].h[0] = *(const v8bf*)(ar);
            af[i].h[1] = *(const v8bf*)(ar + 16);
        }
        V16U bfv[4];
        for (int j = 0; j < 4; ++j) {
            const bf16_t* br = &smem[8192 + abase + (wn + j * 16 + l16) * 32 + half * 8];
            bfv[j].h[0] = *(const v8bf*)(br);
            bfv[j].h[1] = *(const v8bf*)(br + 16);
        }
        for (int i = 0; i < 2; ++i)
            for (int j = 0; j < 4; ++j)
                acc[i][j] = __builtin_amdgcn_wmma_f32_16x16x32_bf16(
                    false, af[i].v, false, bfv[j].v,
                    (short)0, acc[i][j], false, false);
        __syncthreads();        // buffer cur free for slab s+2 staging
    }

    const float bscale = (MODE == 6 && scalarp) ? scalarp[0] : 0.0f;

    if (MODE == 2 || MODE == 3) {
        // Whole block maps to one (b,h): transpose tile in LDS, then write
        // coalesced B128 runs along t into Obf[b][h][d][t].
        for (int i = 0; i < 2; ++i)
            for (int j = 0; j < 4; ++j)
                for (int r = 0; r < 8; ++r) {
                    const int ml = wm + i * 16 + half * 8 + r;   // t within tile
                    const int nl = wn + j * 16 + l16;            // d within tile
                    const float v = acc[i][j][r];
                    smem[nl * 128 + ml] = (MODE == 2) ? f2bf(phif(v)) : f2bf(v);
                }
        __syncthreads();
        const int b  = m0 >> 12, t0 = m0 & 4095;                 // T = 4096
        const int h  = n0 >> 7;                                  // D = 128
        const int dr = tid >> 1;            // d row 0..127
        const int ts = (tid & 1) << 6;      // 64-element t segment
        bf16_t* dst = Obf + (((long)(b * 16 + h) * 128 + dr) << 12) + t0 + ts;
        const bf16_t* srow = &smem[dr * 128 + ts];
        for (int q = 0; q < 8; ++q)
            *(v8bf*)(dst + q * 8) = *(const v8bf*)(srow + q * 8);
    } else {
        // C/D layout: VGPR r -> M = r + 8*half, N = lane&15 (per 16x16 frag)
        for (int i = 0; i < 2; ++i)
            for (int j = 0; j < 4; ++j)
                for (int r = 0; r < 8; ++r) {
                    const int m = m0 + wm + i * 16 + half * 8 + r;
                    const int n = n0 + wn + j * 16 + l16;
                    const float v = acc[i][j][r];
                    if (MODE == 1) {
                        Obf[o_off + (long)m * ldo + n] = f2bf(phif(v));
                    } else if (MODE == 4) {
                        const float sc = 1.0f / fmaxf(aux0[m], 1e-6f);
                        Obf[o_off + (long)m * ldo + n] = f2bf(v * sc);
                    } else if (MODE == 5) {
                        Of32[o_off + (long)m * ldo + n] = aux0[(long)m * ldo + n] + v;
                    } else if (MODE == 6) {
                        Of32[(long)m * ldo + n] = aux0[(long)m * ldo + n] + bscale * v;
                    }
                }
    }
}

// ---------------------------------------------------------------------------
// Small helper kernels
// ---------------------------------------------------------------------------
__global__ void cvt_f32_bf16(bf16_t* __restrict__ dst, const float* __restrict__ src, long n) {
    long i = (long)blockIdx.x * blockDim.x + threadIdx.x;
    if (i < n) dst[i] = f2bf(src[i]);
}

// Mt[h][e][d] = bf16(M[h][d][e])  (transposed per head for retrieve B-operand)
__global__ void cvt_M_transpose(bf16_t* __restrict__ dst, const float* __restrict__ src) {
    int i = blockIdx.x * blockDim.x + threadIdx.x;
    if (i < 16 * 128 * 128) {
        int h = i >> 14, r = i & 16383;
        int d = r >> 7, e = r & 127;
        dst[(h << 14) + (e << 7) + d] = f2bf(src[i]);
    }
}

// norm[b][h][t] = dot(phiQ[b][t][h*128:...], z[h][:])
__global__ void norm_kernel(const bf16_t* __restrict__ phiQ, const float* __restrict__ z,
                            float* __restrict__ norm) {
    int idx = blockIdx.x * blockDim.x + threadIdx.x;     // 0..B*H*T-1
    if (idx >= 2 * 16 * 4096) return;
    int t = idx & 4095;
    int h = (idx >> 12) & 15;
    int b = idx >> 16;
    const bf16_t* row = phiQ + ((long)(b * 4096 + t)) * 2048 + h * 128;
    const float*  zr  = z + h * 128;
    float s = 0.0f;
    #pragma unroll 8
    for (int d = 0; d < 128; ++d) s += (float)row[d] * zr[d];
    norm[idx] = s;
}

// dst[i] = 0.5*(src[i] + src[i+stride])   (mean over batch B=2)
__global__ void avg2_kernel(bf16_t* __restrict__ dst, const bf16_t* __restrict__ src,
                            long n, long stride) {
    long i = (long)blockIdx.x * blockDim.x + threadIdx.x;
    if (i < n) dst[i] = f2bf(0.5f * ((float)src[i] + (float)src[i + stride]));
}

// z_new[h*128+d] = z[h*128+d] + sum_t phiKT_avg[(h*128+d)*T + t]
__global__ void znew_kernel(const bf16_t* __restrict__ phiKT_avg,
                            const float* __restrict__ zin, float* __restrict__ zout) {
    int idx = blockIdx.x * blockDim.x + threadIdx.x;
    if (idx >= 2048) return;
    const bf16_t* row = phiKT_avg + (long)idx * 4096;
    float s = 0.0f;
    #pragma unroll 8
    for (int t = 0; t < 4096; ++t) s += (float)row[t];
    zout[idx] = zin[idx] + s;
}

__global__ void beta_kernel(const float* __restrict__ beta, float* __restrict__ s) {
    if (threadIdx.x == 0) {
        float a = 0.0f;
        for (int i = 0; i < 16; ++i) a += 1.0f / (1.0f + __expf(-beta[i]));
        s[0] = a * (1.0f / 16.0f);
    }
}

// ---------------------------------------------------------------------------
// Launcher
// ---------------------------------------------------------------------------
extern "C" void kernel_launch(void* const* d_in, const int* in_sizes, int n_in,
                              void* d_out, int out_size, void* d_ws, size_t ws_size,
                              hipStream_t stream) {
    (void)in_sizes; (void)n_in; (void)out_size; (void)ws_size;

    const float* hs   = (const float*)d_in[0];   // (B,T,HID)   = 2*4096*2048
    const float* Wq   = (const float*)d_in[1];   // (2048,2048)
    const float* Wk   = (const float*)d_in[2];
    const float* Wv   = (const float*)d_in[3];
    const float* Wout = (const float*)d_in[4];
    const float* beta = (const float*)d_in[5];   // (16,)
    const float* Min  = (const float*)d_in[6];   // (16,128,128)
    const float* zin  = (const float*)d_in[7];   // (16,128)

    float* out_mixed = (float*)d_out;                       // 16777216 f32
    float* out_M     = out_mixed + 16777216;                // 262144 f32
    float* out_z     = out_M + 262144;                      // 2048 f32

    // workspace carve-up (bytes, 256-aligned)
    char* ws = (char*)d_ws;
    size_t off = 0;
    auto carve = [&](size_t bytes) -> char* {
        char* p = ws + off;
        off += (bytes + 255) & ~(size_t)255;
        return p;
    };
    bf16_t* hs_bf     = (bf16_t*)carve(16777216ull * 2);    // also reused as mem_flat
    bf16_t* Wq_bf     = (bf16_t*)carve(4194304ull * 2);
    bf16_t* Wk_bf     = (bf16_t*)carve(4194304ull * 2);
    bf16_t* Wv_bf     = (bf16_t*)carve(4194304ull * 2);
    bf16_t* Wout_bf   = (bf16_t*)carve(4194304ull * 2);
    bf16_t* phiQ      = (bf16_t*)carve(16777216ull * 2);    // [b][t][h*128+d]
    bf16_t* phiKT     = (bf16_t*)carve(16777216ull * 2);    // [b][h][d][t]
    bf16_t* VT        = (bf16_t*)carve(16777216ull * 2);    // [b][h][d][t]
    bf16_t* phiKT_avg = (bf16_t*)carve(8388608ull * 2);     // [h][d][t]
    bf16_t* VT_avg    = (bf16_t*)carve(8388608ull * 2);     // [h][d][t]
    bf16_t* Mt_bf     = (bf16_t*)carve(262144ull * 2);      // [h][e][d]
    float*  norm      = (float*)carve(131072ull * 4);       // [b][h][t]
    float*  scalar    = (float*)carve(256);
    bf16_t* memflat   = hs_bf;  // reuse: hs_bf is dead after the 3 projections

    // 1) precision conversion (compute-bound GEMMs -> bf16 WMMA path)
    cvt_f32_bf16<<<16777216 / 256, 256, 0, stream>>>(hs_bf, hs, 16777216);
    cvt_f32_bf16<<<4194304 / 256, 256, 0, stream>>>(Wq_bf,   Wq,   4194304);
    cvt_f32_bf16<<<4194304 / 256, 256, 0, stream>>>(Wk_bf,   Wk,   4194304);
    cvt_f32_bf16<<<4194304 / 256, 256, 0, stream>>>(Wv_bf,   Wv,   4194304);
    cvt_f32_bf16<<<4194304 / 256, 256, 0, stream>>>(Wout_bf, Wout, 4194304);
    cvt_M_transpose<<<262144 / 256, 256, 0, stream>>>(Mt_bf, Min);
    beta_kernel<<<1, 32, 0, stream>>>(beta, scalar);

    // 2) projections: (8192 x 2048) @ (2048 x 2048)^T, fused phi / transpose
    dim3 gproj(16, 64, 1);
    gemm_wmma_bf16<1><<<gproj, 256, 0, stream>>>(
        hs_bf, 2048, 0, 0, Wq_bf, 2048, 0, 0, 2048,
        phiQ, nullptr, 2048, 0, 0, nullptr, 0, 0, nullptr);
    gemm_wmma_bf16<2><<<gproj, 256, 0, stream>>>(
        hs_bf, 2048, 0, 0, Wk_bf, 2048, 0, 0, 2048,
        phiKT, nullptr, 0, 0, 0, nullptr, 0, 0, nullptr);
    gemm_wmma_bf16<3><<<gproj, 256, 0, stream>>>(
        hs_bf, 2048, 0, 0, Wv_bf, 2048, 0, 0, 2048,
        VT, nullptr, 0, 0, 0, nullptr, 0, 0, nullptr);

    // 3) norm + batch means
    norm_kernel<<<131072 / 256, 256, 0, stream>>>(phiQ, zin, norm);
    avg2_kernel<<<8388608 / 256, 256, 0, stream>>>(phiKT_avg, phiKT, 8388608, 8388608);
    avg2_kernel<<<8388608 / 256, 256, 0, stream>>>(VT_avg,    VT,    8388608, 8388608);

    // 4) retrieve: per (b,h), (4096 x 128) @ (128 x 128)^T, fused /norm
    dim3 gret(1, 32, 32);   // z = b*16 + h
    gemm_wmma_bf16<4><<<gret, 256, 0, stream>>>(
        phiQ, 2048, 8388608, 128,        // A: + b*T*2048 + h*128
        Mt_bf, 128, 0, 16384,            // B: + h*128*128
        128,
        memflat, nullptr, 2048, 8388608, 128,   // O: + b*T*2048 + h*128
        norm, 65536, 4096,               // aux: + z*4096
        nullptr);

    // 5) memory update: per h, (128 x 4096) @ (4096 x 128) -> M_new = M + ...
    dim3 gupd(1, 1, 16);
    gemm_wmma_bf16<5><<<gupd, 256, 0, stream>>>(
        phiKT_avg, 4096, 0, 524288,      // A: + h*D*T
        VT_avg,    4096, 0, 524288,      // B: + h*D*T
        4096,
        nullptr, out_M, 128, 0, 16384,   // O: + h*D*D
        Min, 0, 16384,                   // aux: + h*D*D
        nullptr);

    // 6) z_new
    znew_kernel<<<8, 256, 0, stream>>>(phiKT_avg, zin, out_z);

    // 7) output projection + residual mix: mixed = hs + sigmoid(beta).mean()*mem_out
    gemm_wmma_bf16<6><<<gproj, 256, 0, stream>>>(
        memflat, 2048, 0, 0, Wout_bf, 2048, 0, 0, 2048,
        nullptr, out_mixed, 2048, 0, 0,
        hs, 0, 0, scalar);
}
